// TopKGate_60026462929317
// MI455X (gfx1250) — matrix operand order repaired and checked by reading, not compiled, not run on hardware
//
#include <hip/hip_runtime.h>
#include <hip/hip_bf16.h>

typedef __attribute__((ext_vector_type(16))) __bf16 v16bf;
typedef __attribute__((ext_vector_type(8)))  float  v8f;

#define HDIM   4096
#define NEXP   64
#define KSEL   6
#define NGRP   8
#define EPG    8      // experts per group
#define TGRP   3      // top groups kept
#define TOK    16384  // total tokens
#define SEQ    4096
#define NBATCH 4

// ---------------------------------------------------------------------------
// ws layout (floats): [0..255]   cnt[b*64+e]   expert selection counts
//                     [256..511] ssum[b*64+e]  softmax score sums over seq
// ---------------------------------------------------------------------------

__global__ void zero_ws_kernel(float* ws) {
    int i = blockIdx.x * blockDim.x + threadIdx.x;
    if (i < 512) ws[i] = 0.0f;
}

// One block = 128 tokens (8 waves x 16 tokens). Each wave computes a
// 16-token x 64-expert logits tile with 4 bf16 WMMAs per K-step.
__global__ __launch_bounds__(256) void router_kernel(
    const float* __restrict__ x,      // [TOK, HDIM]
    const float* __restrict__ gw,     // [NEXP, HDIM]
    float* __restrict__ out_idx,      // [TOK, 6] (indices stored as float)
    float* __restrict__ out_w,        // [TOK, 6]
    float* __restrict__ ws_cnt,       // [NBATCH*64]
    float* __restrict__ ws_ssum)      // [NBATCH*64]
{
    __shared__ float lds_scores[128 * NEXP];  // 32 KB: per-token scores
    __shared__ float lds_cnt[NEXP];

    const int lane = threadIdx.x & 31;
    const int wave = threadIdx.x >> 5;
    const int tok0 = blockIdx.x * 128 + wave * 16;

    if (threadIdx.x < NEXP) lds_cnt[threadIdx.x] = 0.0f;

    // ---------------- WMMA GEMM: logits[16 x 64] per wave ----------------
    v8f acc0 = {}, acc1 = {}, acc2 = {}, acc3 = {};
    const int m  = lane & 15;   // token row within tile / expert col within tile
    const int hi = lane >> 4;   // K-half selector
    const float* xrow  = x  + (size_t)(tok0 + m) * HDIM;
    const float* wbase = gw + (size_t)m * HDIM;   // expert row m of each 16-tile

    for (int kb = 0; kb < HDIM; kb += 32) {
        // A fragment: 16-bit A 16x32 layout — lane (m, hi): K = hi*16+j (contiguous, matches B)
        v16bf a;
        #pragma unroll
        for (int j = 0; j < 8; ++j) {
            a[j]     = (__bf16)xrow[kb + hi * 16 + j];
            a[j + 8] = (__bf16)xrow[kb + hi * 16 + 8 + j];
        }
        // B fragments: 32x16, lane (n=m, hi): K = hi*16 + j, expert = g*16 + n
        const float* wp = wbase + kb + hi * 16;
        v16bf b0, b1, b2, b3;
        #pragma unroll
        for (int j = 0; j < 16; ++j) {
            b0[j] = (__bf16)wp[j];
            b1[j] = (__bf16)wp[(size_t)16 * HDIM + j];
            b2[j] = (__bf16)wp[(size_t)32 * HDIM + j];
            b3[j] = (__bf16)wp[(size_t)48 * HDIM + j];
        }
        acc0 = __builtin_amdgcn_wmma_f32_16x16x32_bf16(false, a, false, b0, (short)0, acc0, false, false);
        acc1 = __builtin_amdgcn_wmma_f32_16x16x32_bf16(false, a, false, b1, (short)0, acc1, false, false);
        acc2 = __builtin_amdgcn_wmma_f32_16x16x32_bf16(false, a, false, b2, (short)0, acc2, false, false);
        acc3 = __builtin_amdgcn_wmma_f32_16x16x32_bf16(false, a, false, b3, (short)0, acc3, false, false);
    }

    // Stage C tiles into LDS as [token][expert].
    // C layout: VGPR r, lanes 0-15 -> M=r, lanes 16-31 -> M=r+8; N = lane&15.
    {
        float* sb = &lds_scores[wave * 16 * NEXP];
        const int col  = lane & 15;
        const int rofs = (lane >> 4) * 8;
        #pragma unroll
        for (int r = 0; r < 8; ++r) {
            const int t = r + rofs;
            sb[t * NEXP +  0 + col] = acc0[r];
            sb[t * NEXP + 16 + col] = acc1[r];
            sb[t * NEXP + 32 + col] = acc2[r];
            sb[t * NEXP + 48 + col] = acc3[r];
        }
    }
    __syncthreads();

    // ---------------- softmax + group-limited top-k (1 lane / token) -----
    if (lane < 16) {
        const int tok = tok0 + lane;
        float* sc = &lds_scores[(wave * 16 + lane) * NEXP];

        float mx = -3.0e38f;
        for (int e = 0; e < NEXP; ++e) mx = fmaxf(mx, sc[e]);
        float sum = 0.0f;
        for (int e = 0; e < NEXP; ++e) { float v = __expf(sc[e] - mx); sc[e] = v; sum += v; }
        const float inv = 1.0f / sum;
        for (int e = 0; e < NEXP; ++e) sc[e] *= inv;   // normalized scores in LDS

        // group maxima (8 groups of 8)
        float gm[NGRP];
        for (int g = 0; g < NGRP; ++g) {
            float v = sc[g * EPG];
            for (int j = 1; j < EPG; ++j) v = fmaxf(v, sc[g * EPG + j]);
            gm[g] = v;
        }
        // top-3 groups (ties -> lowest index, matching top_k order)
        unsigned allowed = 0;
        for (int t = 0; t < TGRP; ++t) {
            int bi = 0; float bv = -1.0f;
            for (int g = 0; g < NGRP; ++g)
                if (!((allowed >> g) & 1u) && gm[g] > bv) { bv = gm[g]; bi = g; }
            allowed |= 1u << bi;
        }
        // top-6 experts within allowed groups (descending)
        unsigned long long sel = 0ull;
        int   idxs[KSEL];
        float vals[KSEL];
        float wsum = 0.0f;
        for (int t = 0; t < KSEL; ++t) {
            int bi = 0; float bv = -1.0f;
            for (int e = 0; e < NEXP; ++e) {
                if (!((allowed >> (e >> 3)) & 1u)) continue;
                if ((sel >> e) & 1ull) continue;
                const float v = sc[e];
                if (v > bv) { bv = v; bi = e; }
            }
            sel |= 1ull << bi;
            idxs[t] = bi; vals[t] = bv; wsum += bv;
        }
        const float winv = 1.0f / (wsum + 1e-20f);
        for (int t = 0; t < KSEL; ++t) {
            out_idx[(size_t)tok * KSEL + t] = (float)idxs[t];
            out_w  [(size_t)tok * KSEL + t] = vals[t] * winv;
            atomicAdd(&lds_cnt[idxs[t]], 1.0f);
        }
    }
    __syncthreads();

    // ---------------- block reduction of aux-loss statistics -------------
    const int bidx = (blockIdx.x * 128) / SEQ;   // whole block in one batch elem
    if (threadIdx.x < NEXP) {
        const int e = threadIdx.x;
        float s = 0.0f;
        for (int t = 0; t < 128; ++t) s += lds_scores[t * NEXP + e];
        atomicAdd(&ws_ssum[bidx * NEXP + e], s);
        atomicAdd(&ws_cnt [bidx * NEXP + e], lds_cnt[e]);
    }
}

__global__ void finalize_kernel(const float* __restrict__ ws_cnt,
                                const float* __restrict__ ws_ssum,
                                float* __restrict__ aux_out)
{
    __shared__ float red[256];
    const int tid = threadIdx.x;
    // ce = cnt / (S*K/E) = cnt / 384 ; mean score = ssum / S
    float v = (ws_cnt[tid] * (1.0f / 384.0f)) * (ws_ssum[tid] * (1.0f / (float)SEQ));
    red[tid] = v;
    __syncthreads();
    for (int s = 128; s > 0; s >>= 1) {
        if (tid < s) red[tid] += red[tid + s];
        __syncthreads();
    }
    if (tid == 0) aux_out[0] = red[0] * (0.001f / (float)NBATCH);
}

extern "C" void kernel_launch(void* const* d_in, const int* in_sizes, int n_in,
                              void* d_out, int out_size, void* d_ws, size_t ws_size,
                              hipStream_t stream) {
    const float* x  = (const float*)d_in[0];   // [4,4096,4096] -> [16384,4096]
    const float* gw = (const float*)d_in[1];   // [64,4096]
    float* out      = (float*)d_out;
    float* out_idx  = out;                     // [16384*6]
    float* out_w    = out + (size_t)TOK * KSEL;
    float* aux      = out + (size_t)2 * TOK * KSEL;
    float* ws       = (float*)d_ws;

    zero_ws_kernel<<<2, 256, 0, stream>>>(ws);
    router_kernel<<<TOK / 128, 256, 0, stream>>>(x, gw, out_idx, out_w, ws, ws + 256);
    finalize_kernel<<<1, 256, 0, stream>>>(ws, ws + 256, aux);
}
